// LLMNLLMEModel_57277683860076
// MI455X (gfx1250) — matrix operand order repaired and checked by reading, not compiled, not run on hardware
//
#include <hip/hip_runtime.h>
#include <hip/hip_bf16.h>

// ---------------------------------------------------------------------------
// Hypergraph GNN forward for MI455X (gfx1250, wave32).
// GEMMs: v_wmma_f32_16x16x32_bf16, 32x128 tile per workgroup, K-step 64,
// 4 WMMA per loop body per wave, branch-free (mode templated, edge buffers
// padded to a 32-row multiple so stores/epilogues need no guards).
// Irregular incidence traffic (~15 GB @ 23.3 TB/s dominates) uses float4
// gathers + f32 global atomics / encoded-uint atomicMin.
// ---------------------------------------------------------------------------

#define NUM_NODES 100000
#define NUM_EDGES 50000
#define EPAD      50016          // NUM_EDGES rounded up to 32
#define NNZ       1600000
#define IN_C      768
#define HID       128
#define EPS       1e-5f

#define M_LNPROJ  0
#define M_PLAIN   1
#define M_COMBINE 2
#define M_FUSION  3

typedef __bf16 bf16_t;
typedef __attribute__((ext_vector_type(16))) bf16_t v16bf;
typedef __attribute__((ext_vector_type(8)))  float  v8f;

union Frag {
    v16bf v;
    unsigned short s[16];
};

// Native bf16 converts (gfx1250 has hardware bf16 cvt; RTNE by default).
static __device__ __forceinline__ unsigned short f2bf(float x) {
    union { bf16_t h; unsigned short s; } t;
    t.h = (bf16_t)x;
    return t.s;
}
static __device__ __forceinline__ unsigned pack2(float lo, float hi) {
    union { bf16_t h[2]; unsigned u; } t;
    t.h[0] = (bf16_t)lo;
    t.h[1] = (bf16_t)hi;
    return t.u;
}
static __device__ __forceinline__ float lrelu(float x) { return x > 0.f ? x : 0.01f * x; }
static __device__ __forceinline__ unsigned fenc(float f) {
    unsigned u = __float_as_uint(f);
    return (u & 0x80000000u) ? ~u : (u | 0x80000000u);
}
static __device__ __forceinline__ float fdec(unsigned u) {
    unsigned s = (u & 0x80000000u) ? (u & 0x7FFFFFFFu) : ~u;
    return __uint_as_float(s);
}

// ---------------------------------------------------------------------------
// WMMA GEMM:  C[M,128] = epilogue( preproc(A)[M,K] @ W[K,128] )
// block = 256 threads (8 waves). Workgroup tile: 32 rows x 128 cols.
// Wave w owns cols [16w,16w+16); two row sub-tiles -> acc0/acc1.
// K stepped by 64 -> 4 WMMA per loop body. K in {128, 256, 768}.
// Grid covers ceil32(M) rows; input loads clamp row to M-1 (branchless),
// stores always land in (padded) C — caller guarantees C has ceil32(M) rows,
// and aux/rowscale (when used) are padded likewise.
// ---------------------------------------------------------------------------
template <int MODE>
__global__ __launch_bounds__(256) void gemm_wmma(
    const float* __restrict__ A, int ldA, int K, int M,
    const float* __restrict__ W,
    const float* __restrict__ bias,       // [128]
    const float* __restrict__ rowc,       // [M,2] per-row LN consts (LNPROJ)
    const float* __restrict__ gvec,       // [K]  LN gamma (LNPROJ)
    const float* __restrict__ bvec,       // [K]  LN beta  (LNPROJ)
    const float* __restrict__ aux,        // [ceil32(M),128] raw sums (COMBINE)
    const float* __restrict__ rowscale,   // [ceil32(M)]  D^-1 (COMBINE)
    const float* __restrict__ bias2,      // [128] skip bias (COMBINE)
    const unsigned* __restrict__ aggu,    // [.,128] encoded seg-min (FUSION)
    const float* __restrict__ A2,         // [.,128] x_e half (FUSION)
    float* __restrict__ C)
{
    __shared__ unsigned short As[32][68];   // 32 rows x 64 k (+pad)
    __shared__ unsigned short Bs[128][68];  // transposed: [n][k]

    const int tid  = threadIdx.x;
    const int wave = tid >> 5;
    const int lane = tid & 31;
    const int half = lane >> 4;
    const int l15  = lane & 15;
    const int row0 = blockIdx.x * 32;

    v8f acc0 = {};
    v8f acc1 = {};

    for (int kc = 0; kc < K; kc += 64) {
        // ---- stage A tile (32 rows x 64 k) as float4 loads ----
#pragma unroll
        for (int p = 0; p < 2; ++p) {
            int idx  = tid + p * 256;          // 0..511 float4 slots
            int r    = idx >> 4;               // 0..31
            int k4   = (idx & 15) * 4;         // 0..60
            int grow = min(row0 + r, M - 1);   // branchless input clamp
            int gk   = kc + k4;
            float4 v;
            if (MODE == M_FUSION) {
                if (gk < HID) {
                    uint4 u = *(const uint4*)(aggu + (long)grow * HID + gk);
                    v.x = fdec(u.x); v.y = fdec(u.y); v.z = fdec(u.z); v.w = fdec(u.w);
                } else {
                    v = *(const float4*)(A2 + (long)grow * HID + (gk - HID));
                }
            } else {
                v = *(const float4*)(A + (long)grow * ldA + gk);
                if (MODE == M_LNPROJ) {
                    float c0 = rowc[2 * grow], c1 = rowc[2 * grow + 1];
                    v.x = fmaf(fmaf(v.x, c1, c0), gvec[gk + 0], bvec[gk + 0]);
                    v.y = fmaf(fmaf(v.y, c1, c0), gvec[gk + 1], bvec[gk + 1]);
                    v.z = fmaf(fmaf(v.z, c1, c0), gvec[gk + 2], bvec[gk + 2]);
                    v.w = fmaf(fmaf(v.w, c1, c0), gvec[gk + 3], bvec[gk + 3]);
                }
            }
            unsigned* dstp = (unsigned*)&As[r][k4];
            dstp[0] = pack2(v.x, v.y);
            dstp[1] = pack2(v.z, v.w);
        }
        // ---- stage B tile (64 k x 128 n), float4 reads, transposed writes ----
#pragma unroll
        for (int p = 0; p < 8; ++p) {
            int idx = tid + p * 256;           // 0..2047 float4 slots
            int k   = idx >> 5;                // 0..63
            int n4  = (idx & 31) * 4;          // 0..124
            float4 v = *(const float4*)(W + (long)(kc + k) * HID + n4);
            Bs[n4 + 0][k] = f2bf(v.x);
            Bs[n4 + 1][k] = f2bf(v.y);
            Bs[n4 + 2][k] = f2bf(v.z);
            Bs[n4 + 3][k] = f2bf(v.w);
        }
        __syncthreads();

        // ---- 2 k-substeps x 2 row-tiles = 4 WMMA ----
#pragma unroll
        for (int ks = 0; ks < 64; ks += 32) {
            Frag b, a0, a1;
#pragma unroll
            for (int r = 0; r < 8; ++r) {
                int j  = 2 * r;
                int ka = ks + j + 8 * (((j >= 8) ? 1 : 0) + half);  // A 16x32 layout
                int kb = ks + j + 8 * (((j >= 8) ? 1 : 0) + half);  // B 32x16 layout
                a0.s[j]     = As[l15][ka];
                a0.s[j + 1] = As[l15][ka + 1];
                a1.s[j]     = As[16 + l15][ka];
                a1.s[j + 1] = As[16 + l15][ka + 1];
                b.s[j]      = Bs[wave * 16 + l15][kb];
                b.s[j + 1]  = Bs[wave * 16 + l15][kb + 1];
            }
            acc0 = __builtin_amdgcn_wmma_f32_16x16x32_bf16(
                false, a0.v, false, b.v, (short)0, acc0, false, false);
            acc1 = __builtin_amdgcn_wmma_f32_16x16x32_bf16(
                false, a1.v, false, b.v, (short)0, acc1, false, false);
        }
        __syncthreads();
    }

    // ---- epilogue (no guards: C/aux/rowscale are padded to ceil32(M)) ----
    // C layout: lanes 0-15 -> M=i, lanes 16-31 -> M=i+8.
    const int col = wave * 16 + l15;
    const float bcol  = (MODE == M_PLAIN) ? 0.f : bias[col];
    const float b2col = (MODE == M_COMBINE) ? bias2[col] : 0.f;
#pragma unroll
    for (int rt = 0; rt < 2; ++rt) {
        v8f acc = rt ? acc1 : acc0;
        const int rbase = row0 + rt * 16 + 8 * half;
        if (MODE == M_COMBINE) {
            float av[8], rv[8];
#pragma unroll
            for (int i = 0; i < 8; ++i) {
                av[i] = aux[(long)(rbase + i) * HID + col];
                rv[i] = rowscale[rbase + i];
            }
#pragma unroll
            for (int i = 0; i < 8; ++i) {
                float y = lrelu(fmaf(av[i], rv[i], bcol)) + acc[i] + b2col;
                C[(long)(rbase + i) * HID + col] = y;
            }
        } else {
#pragma unroll
            for (int i = 0; i < 8; ++i) {
                float y = acc[i];
                if (MODE == M_LNPROJ)      y = lrelu(y + bcol);
                else if (MODE == M_FUSION) y = y + bcol;
                C[(long)(rbase + i) * HID + col] = y;
            }
        }
    }
}

// ---------------------------------------------------------------------------
// Per-row stats: c0 = -mu*rs, c1 = inv*rs so that LN(v) = (v*c1 + c0).
// L2 pre-normalization folds in (LN is scale-invariant up to EPS).
// ---------------------------------------------------------------------------
__global__ void rowstats(const float* __restrict__ A, float* __restrict__ st,
                         int M, int K, int l2) {
    int row  = blockIdx.x * (blockDim.x >> 5) + (threadIdx.x >> 5);
    int lane = threadIdx.x & 31;
    if (row >= M) return;
    const float* p = A + (long)row * K;
    float s = 0.f, q = 0.f;
    for (int k = lane; k < K; k += 32) { float v = p[k]; s += v; q += v * v; }
#pragma unroll
    for (int m = 16; m >= 1; m >>= 1) { s += __shfl_xor(s, m, 32); q += __shfl_xor(q, m, 32); }
    if (lane == 0) {
        float inv = 1.0f;
        if (l2) { float n2 = fmaxf(sqrtf(q), 1e-12f); inv = 1.0f / n2; }
        float mu  = s * inv / (float)K;
        float var = q * inv * inv / (float)K - mu * mu;
        float rs  = rsqrtf(fmaxf(var, 0.f) + EPS);
        st[2 * row]     = -mu * rs;
        st[2 * row + 1] = inv * rs;
    }
}

// GraphNorm column stats (sum, sumsq) via block-local accumulation + atomics.
__global__ void colstats(const float* __restrict__ H, float* __restrict__ csum,
                         float* __restrict__ csq, int M) {
    int f = threadIdx.x;  // 128
    float s = 0.f, q = 0.f;
    for (int r = blockIdx.x; r < M; r += gridDim.x) {
        float v = H[(long)r * HID + f];
        s += v; q += v * v;
    }
    atomicAdd(&csum[f], s);
    atomicAdd(&csq[f], q);
}

// Fold GraphNorm into per-feature affine: y = v*s1[f] + s0[f].
__global__ void colfinal(const float* __restrict__ csum, const float* __restrict__ csq,
                         const float* __restrict__ w, const float* __restrict__ b,
                         const float* __restrict__ ms,
                         float* __restrict__ s0, float* __restrict__ s1, float invM) {
    int f = threadIdx.x;
    float mu  = csum[f] * invM;
    float e2  = csq[f] * invM;
    float m   = ms[f];
    float var = e2 - 2.f * m * mu * mu + m * m * mu * mu;  // E[(v-ms*mu)^2]
    float rs  = rsqrtf(fmaxf(var, 0.f) + EPS);
    float S1  = w[f] * rs;
    s1[f] = S1;
    s0[f] = b[f] - m * mu * S1;
}

__global__ void applyaff(float* __restrict__ H, const float* __restrict__ s0,
                         const float* __restrict__ s1, long n4) {
    long i = blockIdx.x * (long)blockDim.x + threadIdx.x;
    if (i >= n4) return;
    int f = (int)((i * 4) & (HID - 1));
    float4 v = ((float4*)H)[i];
    v.x = fmaf(v.x, s1[f + 0], s0[f + 0]);
    v.y = fmaf(v.y, s1[f + 1], s0[f + 1]);
    v.z = fmaf(v.z, s1[f + 2], s0[f + 2]);
    v.w = fmaf(v.w, s1[f + 3], s0[f + 3]);
    ((float4*)H)[i] = v;
}

// out[sidx[e]] += in[gidx[e]] * (HAS_SCALE ? scale[gidx[e]] : 1)
// 32 lanes per incidence, float4 per lane (128 features).
template <bool HAS_SCALE>
__global__ void scatter_add(float* __restrict__ out, const float* __restrict__ in,
                            const int* __restrict__ gidx, const int* __restrict__ sidx,
                            const float* __restrict__ scale, long nnz) {
    long t = blockIdx.x * (long)blockDim.x + threadIdx.x;
    long e = t >> 5;
    if (e >= nnz) return;
    int f  = ((int)t & 31) * 4;
    int gr = gidx[e], sr = sidx[e];
    float4 v = *(const float4*)(in + (long)gr * HID + f);
    float  k = HAS_SCALE ? scale[gr] : 1.0f;
    float* o = out + (long)sr * HID + f;
    atomicAdd(o + 0, v.x * k);
    atomicAdd(o + 1, v.y * k);
    atomicAdd(o + 2, v.z * k);
    atomicAdd(o + 3, v.w * k);
}

// segment_min via order-preserving uint encoding + atomicMin.
__global__ void segmin(unsigned* __restrict__ out, const float* __restrict__ in,
                       const int* __restrict__ gidx, const int* __restrict__ sidx,
                       long nnz) {
    long t = blockIdx.x * (long)blockDim.x + threadIdx.x;
    long e = t >> 5;
    if (e >= nnz) return;
    int f  = ((int)t & 31) * 4;
    int gr = gidx[e], sr = sidx[e];
    float4 v = *(const float4*)(in + (long)gr * HID + f);
    unsigned* o = out + (long)sr * HID + f;
    atomicMin(o + 0, fenc(v.x));
    atomicMin(o + 1, fenc(v.y));
    atomicMin(o + 2, fenc(v.z));
    atomicMin(o + 3, fenc(v.w));
}

__global__ void degk(const int* __restrict__ src, const int* __restrict__ dst,
                     float* __restrict__ dn, float* __restrict__ de, long nnz) {
    long t = blockIdx.x * (long)blockDim.x + threadIdx.x;
    if (t >= nnz) return;
    atomicAdd(&dn[src[t]], 1.0f);
    atomicAdd(&de[dst[t]], 1.0f);
}

__global__ void recipk(float* __restrict__ d, int n) {
    int i = blockIdx.x * blockDim.x + threadIdx.x;
    if (i < n) d[i] = d[i] > 0.f ? 1.0f / d[i] : 0.f;
}

// out[row] = dot(H[row,:128], w) + b (wave per row, wave32 shuffle reduce)
__global__ void finaldot(const float* __restrict__ H, const float* __restrict__ w,
                         const float* __restrict__ b, float* __restrict__ out, int M) {
    int row  = blockIdx.x * (blockDim.x >> 5) + (threadIdx.x >> 5);
    int lane = threadIdx.x & 31;
    if (row >= M) return;
    const float* p = H + (long)row * HID;
    float s = 0.f;
    for (int k = lane; k < HID; k += 32) s += p[k] * w[k];
#pragma unroll
    for (int m = 16; m >= 1; m >>= 1) s += __shfl_xor(s, m, 32);
    if (lane == 0) out[row] = s + b[0];
}

// ---------------------------------------------------------------------------
// Host orchestration.
// Parameter leaves assumed in jax tree (sorted-key) flatten order:
//  3:e_norm.b 4:e_norm.g 5:e_proj.W 6:e_proj.b 7:fusion.W 8:fusion.b
//  9:in_norm.b 10:in_norm.g 11:in_proj.W 12:in_proj.b
//  13+14L: gn.b gn.ms gn.w gn_d.b gn_d.ms gn_d.w hg.W hg.b hg_d.W hg_d.b
//          skip.W skip.b skip_d.W skip_d.b
//  41:linear.W 42:linear.b
// ---------------------------------------------------------------------------
extern "C" void kernel_launch(void* const* d_in, const int* in_sizes, int n_in,
                              void* d_out, int out_size, void* d_ws, size_t ws_size,
                              hipStream_t stream) {
    (void)in_sizes; (void)n_in; (void)out_size; (void)ws_size;

    const float* x    = (const float*)d_in[0];
    const float* x_e  = (const float*)d_in[1];
    const int*   ei   = (const int*)d_in[2];
    const int*   src  = ei;
    const int*   dst  = ei + NNZ;
    auto P = [&](int i) { return (const float*)d_in[i]; };

    const size_t NB  = (size_t)NUM_NODES * HID * sizeof(float);  // 51.2 MB
    const size_t EBP = (size_t)EPAD * HID * sizeof(float);       // padded edge buf

    size_t off = 0;
    auto alloc = [&](size_t bytes) {
        void* p = (char*)d_ws + off;
        off = (off + bytes + 255) & ~(size_t)255;
        return p;
    };
    float* N0 = (float*)alloc(NB);
    float* N1 = (float*)alloc(NB);
    float* N2 = (float*)alloc(NB);
    float* E0 = (float*)alloc(EBP);
    float* E1 = (float*)alloc(EBP);
    float* E2 = (float*)alloc(EBP);
    float* stats_x = (float*)alloc((size_t)NUM_NODES * 2 * sizeof(float));
    float* stats_e = (float*)alloc((size_t)NUM_EDGES * 2 * sizeof(float));
    float* deg_n   = (float*)alloc((size_t)NUM_NODES * sizeof(float));
    float* deg_e   = (float*)alloc((size_t)EPAD * sizeof(float));  // padded
    float* colbuf  = (float*)alloc(4 * HID * sizeof(float));
    float* csum = colbuf, * csq = colbuf + HID, * s0 = colbuf + 2 * HID, * s1 = colbuf + 3 * HID;

    const dim3 B256(256);
    const int scatterBlocks = (int)(((long)NNZ * 32 + 255) / 256);
    const int gN = NUM_NODES / 32;   // 3125 (exact)
    const int gE = EPAD / 32;        // 1563 (covers padded edge rows)

    // ---- degrees -> reciprocals (deterministic, recomputed each call) ----
    hipMemsetAsync(deg_n, 0, (size_t)NUM_NODES * sizeof(float), stream);
    hipMemsetAsync(deg_e, 0, (size_t)EPAD * sizeof(float), stream);
    degk<<<(NNZ + 255) / 256, B256, 0, stream>>>(src, dst, deg_n, deg_e, NNZ);
    recipk<<<(NUM_NODES + 255) / 256, B256, 0, stream>>>(deg_n, NUM_NODES);
    recipk<<<(NUM_EDGES + 255) / 256, B256, 0, stream>>>(deg_e, NUM_EDGES);

    // ---- input projections (LN fused into A-loader) ----
    rowstats<<<(NUM_NODES + 7) / 8, B256, 0, stream>>>(x, stats_x, NUM_NODES, IN_C, 0);
    rowstats<<<(NUM_EDGES + 7) / 8, B256, 0, stream>>>(x_e, stats_e, NUM_EDGES, IN_C, 1);
    gemm_wmma<M_LNPROJ><<<gN, B256, 0, stream>>>(x, IN_C, IN_C, NUM_NODES,
        P(11), P(12), stats_x, P(10), P(9),
        nullptr, nullptr, nullptr, nullptr, nullptr, N0);
    gemm_wmma<M_LNPROJ><<<gE, B256, 0, stream>>>(x_e, IN_C, IN_C, NUM_EDGES,
        P(5), P(6), stats_e, P(4), P(3),
        nullptr, nullptr, nullptr, nullptr, nullptr, E0);

    float *hx = N0, *he = E0, *nf0 = N1, *nf1 = N2, *ef0 = E1, *ef1 = E2;

    for (int L = 0; L < 2; ++L) {
        int b0 = 13 + L * 14;
        const float *gn_b = P(b0+0), *gn_ms = P(b0+1), *gn_w = P(b0+2);
        const float *gd_b = P(b0+3), *gd_ms = P(b0+4), *gd_w = P(b0+5);
        const float *hgW  = P(b0+6), *hg_b  = P(b0+7);
        const float *hgdW = P(b0+8), *hgd_b = P(b0+9);
        const float *skW  = P(b0+10), *sk_b = P(b0+11);
        const float *skdW = P(b0+12), *skd_b = P(b0+13);

        // ===== node path =====
        hipMemsetAsync(colbuf, 0, 2 * HID * sizeof(float), stream);
        colstats<<<512, dim3(HID), 0, stream>>>(hx, csum, csq, NUM_NODES);
        colfinal<<<1, dim3(HID), 0, stream>>>(csum, csq, gn_w, gn_b, gn_ms, s0, s1,
                                              1.0f / NUM_NODES);
        applyaff<<<(NUM_NODES * 32 + 255) / 256, B256, 0, stream>>>(hx, s0, s1,
                                                                    (long)NUM_NODES * 32);
        // vw = gn(x) @ hgW
        gemm_wmma<M_PLAIN><<<gN, B256, 0, stream>>>(hx, HID, HID, NUM_NODES, hgW,
            nullptr, nullptr, nullptr, nullptr, nullptr, nullptr, nullptr, nullptr, nullptr, nf0);
        // out_e = seg_sum(vw[src] -> dst)
        hipMemsetAsync(ef0, 0, EBP, stream);
        scatter_add<false><<<scatterBlocks, B256, 0, stream>>>(ef0, nf0, src, dst, nullptr, NNZ);
        // out_n = seg_sum(out_e[dst] * B^-1 -> src)
        hipMemsetAsync(nf1, 0, NB, stream);
        scatter_add<true><<<scatterBlocks, B256, 0, stream>>>(nf1, ef0, dst, src, deg_e, NNZ);
        // x' = lrelu(out_n*D^-1 + hg_b) + gn(x)@skW + sk_b
        gemm_wmma<M_COMBINE><<<gN, B256, 0, stream>>>(hx, HID, HID, NUM_NODES, skW,
            hg_b, nullptr, nullptr, nullptr, nf1, deg_n, sk_b, nullptr, nullptr, nf0);
        { float* t = hx; hx = nf0; nf0 = t; }

        // ===== dual path (incidence roles flipped) =====
        hipMemsetAsync(colbuf, 0, 2 * HID * sizeof(float), stream);
        colstats<<<512, dim3(HID), 0, stream>>>(he, csum, csq, NUM_EDGES);
        colfinal<<<1, dim3(HID), 0, stream>>>(csum, csq, gd_w, gd_b, gd_ms, s0, s1,
                                              1.0f / NUM_EDGES);
        applyaff<<<(NUM_EDGES * 32 + 255) / 256, B256, 0, stream>>>(he, s0, s1,
                                                                    (long)NUM_EDGES * 32);
        gemm_wmma<M_PLAIN><<<gE, B256, 0, stream>>>(he, HID, HID, NUM_EDGES, hgdW,
            nullptr, nullptr, nullptr, nullptr, nullptr, nullptr, nullptr, nullptr, nullptr, ef0);
        // node-indexed hyper sums: seg_sum(vw_e[dst] -> src)
        hipMemsetAsync(nf0, 0, NB, stream);
        scatter_add<false><<<scatterBlocks, B256, 0, stream>>>(nf0, ef0, dst, src, nullptr, NNZ);
        // edge-indexed result: seg_sum(hyper[src] * (1/deg_n) -> dst)
        hipMemsetAsync(ef1, 0, EBP, stream);
        scatter_add<true><<<scatterBlocks, B256, 0, stream>>>(ef1, nf0, src, dst, deg_n, NNZ);
        gemm_wmma<M_COMBINE><<<gE, B256, 0, stream>>>(he, HID, HID, NUM_EDGES, skdW,
            hgd_b, nullptr, nullptr, nullptr, ef1, deg_e, skd_b, nullptr, nullptr, ef0);
        { float* t = he; he = ef0; ef0 = t; }
    }

    // ---- segment_min(x[src] -> dst) with encoded-uint atomicMin ----
    unsigned* aggu = (unsigned*)ef1;
    hipMemsetAsync(aggu, 0xFF, EBP, stream);
    segmin<<<scatterBlocks, B256, 0, stream>>>(aggu, hx, src, dst, NNZ);

    // ---- edge fusion: concat([min_agg, x_e]) @ Wf + bf ----
    gemm_wmma<M_FUSION><<<gE, B256, 0, stream>>>(nullptr, 2 * HID, 2 * HID, NUM_EDGES,
        P(7), P(8), nullptr, nullptr, nullptr, nullptr, nullptr, nullptr, aggu, he, ef0);

    // ---- final linear 128 -> 1 ----
    finaldot<<<(NUM_EDGES + 7) / 8, B256, 0, stream>>>(ef0, P(41), P(42),
                                                       (float*)d_out, NUM_EDGES);
}
